// Mamba2Block_22376779612832
// MI455X (gfx1250) — matrix operand order repaired and checked by reading, not compiled, not run on hardware
//
#include <hip/hip_runtime.h>

// ---------------------------------------------------------------------------
// Mamba2 block forward for MI455X (gfx1250, wave32, WMMA).
// bf16 WMMA (f32 accumulate) for all GEMMs; f32 for state/elementwise.
// Big GEMMs use double-buffered GLOBAL_LOAD_ASYNC_TO_LDS_B128 (ASYNCcnt).
// ---------------------------------------------------------------------------

typedef __bf16 bf16_t;
typedef __attribute__((ext_vector_type(16))) __bf16 v16bf;
typedef __attribute__((ext_vector_type(2)))  __bf16 bf16x2;
typedef __attribute__((ext_vector_type(8)))  float  v8f;

#define NB_B       2
#define L_SEQ      4096
#define NTOK       8192          // B * L
#define DMODEL     2048
#define DINNER     4096
#define NHEADS     64
#define HEADDIM    64
#define DSTATE     64
#define DXBC       4224          // DINNER + 2*DSTATE
#define DPROJ      8448          // 2*DINNER + 2*DSTATE + NHEADS = 8384 used, + 64 unused
#define NCHUNK     64            // L / 64
#define CHUNKT     64

// ------------------------- WMMA fragment helpers ---------------------------
// 16-bit A matrix 16x32 (MxK): lane<16 -> M=lane, K in {0..7,16..23};
// lane>=16 -> M=lane-16, K in {8..15,24..31}. VGPR v holds K pair:
//   k = 2*(v&3) + 8*half + 16*(v>>2)
static __device__ __forceinline__ v16bf ldsA_frag(const bf16_t* p, int ld) {
  const int lane = threadIdx.x & 31;
  const int half = lane >> 4;
  const int m    = lane & 15;
  v16bf r;
#pragma unroll
  for (int v = 0; v < 8; ++v) {
    const int k = 2 * (v & 3) + 8 * half + 16 * (v >> 2);
    bf16x2 t = *(const bf16x2*)(p + m * ld + k);
    r[2 * v]     = t[0];
    r[2 * v + 1] = t[1];
  }
  return r;
}

// 16-bit B matrix 32x16 (KxN): lane<16 -> N=lane, K in {0..7,16..23};
// lane>=16 -> N=lane-16, K in {8..15,24..31}.  LDS layout is [n][k], k contiguous.
static __device__ __forceinline__ v16bf ldsB_frag(const bf16_t* p, int ld) {
  const int lane = threadIdx.x & 31;
  const int half = lane >> 4;
  const int n    = lane & 15;
  v16bf r;
#pragma unroll
  for (int v = 0; v < 8; ++v) {
    const int k = 2 * (v & 3) + 8 * half + 16 * (v >> 2);
    bf16x2 t = *(const bf16x2*)(p + n * ld + k);
    r[2 * v]     = t[0];
    r[2 * v + 1] = t[1];
  }
  return r;
}

static __device__ __forceinline__ v8f wmma_bf16(v16bf a, v16bf b, v8f c) {
  return __builtin_amdgcn_wmma_f32_16x16x32_bf16(false, a, false, b,
                                                 (short)0, c, false, false);
}

// ---------------- async global -> LDS (CDNA5 ASYNCcnt path) ----------------
// Per-lane DMA of 16B from global to LDS; no VGPR data, tracked by ASYNCcnt.
// LDS address = low 32 bits of the generic shared pointer (addr[31:0]).
static __device__ __forceinline__ void async_ld_b128(unsigned lds,
                                                     const void* g) {
  asm volatile("global_load_async_to_lds_b128 %0, %1, off"
               :: "v"(lds), "v"(g)
               : "memory");
}
static __device__ __forceinline__ void wait_async_le4() {
  asm volatile("s_wait_asynccnt 0x4" ::: "memory");
}
static __device__ __forceinline__ void wait_async_0() {
  asm volatile("s_wait_asynccnt 0x0" ::: "memory");
}

// ---------------------------- convert f32->bf16 ----------------------------
__global__ void f32_to_bf16_kernel(const float* __restrict__ src,
                                   bf16_t* __restrict__ dst, unsigned n) {
  unsigned i = blockIdx.x * 256u + threadIdx.x;
  if (i < n) dst[i] = (bf16_t)src[i];
}

// ------------------------- tiled bf16 WMMA GEMM -----------------------------
// C[M,N] (f32) = A[M,K] (bf16, row major) * B[N,K]^T (bf16, row major).
// Block: 256 threads = 8 waves; tile 128x128, BK=32; each wave: 4x2 WMMA tiles.
// Double-buffered async global->LDS staging overlapped with WMMA compute.
__global__ __launch_bounds__(256) void gemm_bf16_wmma_kernel(
    const bf16_t* __restrict__ A, const bf16_t* __restrict__ B,
    float* __restrict__ C, int M, int N, int K) {
  constexpr int BK  = 32;
  constexpr int LDT = BK + 8;               // pad: 80B row stride, conflict-free
  __shared__ bf16_t sA[2][128 * LDT];
  __shared__ bf16_t sB[2][128 * LDT];

  const int tid  = threadIdx.x;
  const int lane = tid & 31;
  const int wave = tid >> 5;
  const int wm   = (wave >> 2) * 64;        // 0 / 64
  const int wn   = (wave & 3) * 32;         // 0 / 32 / 64 / 96
  const int bm0  = blockIdx.y * 128;
  const int bn0  = blockIdx.x * 128;
  const int lr   = tid >> 1;                // 0..127 tile row
  const int lc   = (tid & 1) * 16;          // 0 / 16

  const bf16_t* gA = A + (size_t)(bm0 + lr) * K + lc;
  const bf16_t* gB = B + (size_t)(bn0 + lr) * K + lc;

  // per-buffer LDS byte addresses for this thread's 2x16B staging slots
  unsigned oA[2], oB[2];
  oA[0] = (unsigned)(uintptr_t)&sA[0][lr * LDT + lc];
  oA[1] = (unsigned)(uintptr_t)&sA[1][lr * LDT + lc];
  oB[0] = (unsigned)(uintptr_t)&sB[0][lr * LDT + lc];
  oB[1] = (unsigned)(uintptr_t)&sB[1][lr * LDT + lc];

  v8f acc[4][2] = {};

  const int nk = K / BK;
  // prologue: tile 0 into buffer 0
  async_ld_b128(oA[0], gA);
  async_ld_b128(oA[0] + 16, gA + 8);
  async_ld_b128(oB[0], gB);
  async_ld_b128(oB[0] + 16, gB + 8);

  for (int ki = 0; ki < nk; ++ki) {
    const int cur = ki & 1;
    if (ki + 1 < nk) {
      // issue next tile into the spare buffer (safe: barrier at end of the
      // previous iteration guarantees everyone is done reading it)
      const bf16_t* nA = gA + (size_t)(ki + 1) * BK;
      const bf16_t* nB = gB + (size_t)(ki + 1) * BK;
      async_ld_b128(oA[cur ^ 1], nA);
      async_ld_b128(oA[cur ^ 1] + 16, nA + 8);
      async_ld_b128(oB[cur ^ 1], nB);
      async_ld_b128(oB[cur ^ 1] + 16, nB + 8);
      wait_async_le4();     // async ops retire in order: tile ki has landed
    } else {
      wait_async_0();
    }
    __syncthreads();        // every wave has waited -> all of tile ki in LDS

    v16bf aF[4], bF[2];
#pragma unroll
    for (int i = 0; i < 4; ++i)
      aF[i] = ldsA_frag(&sA[cur][(wm + i * 16) * LDT], LDT);
#pragma unroll
    for (int j = 0; j < 2; ++j)
      bF[j] = ldsB_frag(&sB[cur][(wn + j * 16) * LDT], LDT);
#pragma unroll
    for (int i = 0; i < 4; ++i)
#pragma unroll
      for (int j = 0; j < 2; ++j)
        acc[i][j] = wmma_bf16(aF[i], bF[j], acc[i][j]);

    __syncthreads();        // done reading buffer `cur`; it may be refilled
  }

  // C/D layout: VGPR v -> row (v + 8*half), col = lane&15.
  const int half = lane >> 4;
  const int nn   = lane & 15;
#pragma unroll
  for (int i = 0; i < 4; ++i)
#pragma unroll
    for (int j = 0; j < 2; ++j)
#pragma unroll
      for (int v = 0; v < 8; ++v) {
        const int row = bm0 + wm + i * 16 + v + 8 * half;
        const int col = bn0 + wn + j * 16 + nn;
        C[(size_t)row * N + col] = acc[i][j][v];
      }
}

// ------------------- depthwise causal conv(4) + SiLU -----------------------
__global__ void conv_silu_kernel(const float* __restrict__ zxbcdt,
                                 const float* __restrict__ conv_w,
                                 const float* __restrict__ conv_b,
                                 float* __restrict__ xbc) {
  unsigned idx = blockIdx.x * 256u + threadIdx.x;   // over NTOK * DXBC
  if (idx >= (unsigned)NTOK * DXBC) return;
  const int ch = idx % DXBC;
  const unsigned bt = idx / DXBC;
  const int t = bt % L_SEQ;
  const int b = bt / L_SEQ;
  const float* src = zxbcdt + ((size_t)b * L_SEQ) * DPROJ + DINNER + ch;
  float acc = conv_b[ch];
#pragma unroll
  for (int k = 0; k < 4; ++k) {
    const int tt = t - 3 + k;
    const float v = (tt >= 0) ? src[(size_t)tt * DPROJ] : 0.f;
    acc += conv_w[ch * 4 + k] * v;
  }
  xbc[idx] = acc / (1.f + __expf(-acc));            // SiLU
}

// --------------------- dt softplus + A*dt precompute -----------------------
__global__ void dt_kernel(const float* __restrict__ zxbcdt,
                          const float* __restrict__ dt_bias,
                          const float* __restrict__ A_log,
                          float* __restrict__ dt_soft,
                          float* __restrict__ a_dt) {
  unsigned idx = blockIdx.x * 256u + threadIdx.x;   // over NTOK * NHEADS
  if (idx >= (unsigned)NTOK * NHEADS) return;
  const int h = idx % NHEADS;
  const unsigned bt = idx / NHEADS;
  const float v = zxbcdt[(size_t)bt * DPROJ + (DINNER + DXBC) + h] + dt_bias[h];
  const float sp = (v > 20.f) ? v : __logf(1.f + __expf(v));
  dt_soft[idx] = sp;
  a_dt[idx] = -__expf(A_log[h]) * sp;
}

// ---------------- intra-chunk SSD: CB, Y_diag, local state -----------------
// One workgroup (128 thr = 4 waves) per (b, chunk, head). T=P=N=64.
__global__ __launch_bounds__(128) void scan_intra_kernel(
    const float* __restrict__ xbc, const float* __restrict__ dt_s,
    const float* __restrict__ a_dt, float* __restrict__ y,
    float* __restrict__ states) {
  constexpr int LD = 72;
  __shared__ bf16_t sC[64 * LD];    // C  [t][n]   (A of CB, k=n)
  __shared__ bf16_t sB[64 * LD];    // B  [s][n]   (B of CB, col=s, k=n)
  __shared__ bf16_t sBT[64 * LD];   // B^T[n][t]   (B of state, col=n, k=t)
  __shared__ bf16_t sXT[64 * LD];   // X^T[p][t]   (B of Ydiag, col=p, k=s)
  __shared__ bf16_t sXD[64 * LD];   // (decay*X)^T [p][t] (A of state, k=t)
  __shared__ bf16_t sM[64 * LD];    // L∘CB [t][s] (A of Ydiag, k=s)
  __shared__ float  sAcum[64], sDT[64];

  const int bx = blockIdx.x;
  const int h = bx & 63, c = (bx >> 6) & 63, b = bx >> 12;
  const int tid = threadIdx.x, lane = tid & 31, wave = tid >> 5;
  const size_t tok0 = (size_t)b * L_SEQ + c * CHUNKT;

  if (tid < 64) {
    sAcum[tid] = a_dt[(tok0 + tid) * NHEADS + h];
    sDT[tid]   = dt_s[(tok0 + tid) * NHEADS + h];
  }
  __syncthreads();
  if (tid == 0) {                       // inclusive cumsum of a = A*dt
    float s = 0.f;
    for (int t = 0; t < 64; ++t) { s += sAcum[t]; sAcum[t] = s; }
  }
  __syncthreads();
  const float alast = sAcum[63];

  for (int i = tid; i < 64 * 64; i += 128) {
    const int t = i >> 6, n = i & 63;
    const float* row = xbc + (tok0 + t) * DXBC;
    const float cv = row[DINNER + DSTATE + n];
    const float bv = row[DINNER + n];
    sC[t * LD + n]  = (bf16_t)cv;
    sB[t * LD + n]  = (bf16_t)bv;
    sBT[n * LD + t] = (bf16_t)bv;
    const float xv = row[h * HEADDIM + n] * sDT[t];   // X = x * dt (n == p)
    sXT[n * LD + t] = (bf16_t)xv;
    sXD[n * LD + t] = (bf16_t)(xv * __expf(alast - sAcum[t]));
  }
  __syncthreads();

  const int m0 = wave * 16;
  const int half = lane >> 4, nn = lane & 15;

  // GEMM 1: CB[t,s] = sum_n C[t,n] B[s,n]; then M = CB ∘ Lmat -> sM (bf16)
  {
    v8f acc[4] = {};
#pragma unroll
    for (int kk = 0; kk < 64; kk += 32) {
      v16bf aF = ldsA_frag(&sC[m0 * LD + kk], LD);
#pragma unroll
      for (int j = 0; j < 4; ++j)
        acc[j] = wmma_bf16(aF, ldsB_frag(&sB[(j * 16) * LD + kk], LD), acc[j]);
    }
#pragma unroll
    for (int j = 0; j < 4; ++j)
#pragma unroll
      for (int v = 0; v < 8; ++v) {
        const int t = m0 + v + 8 * half;
        const int s = j * 16 + nn;
        const float lm = (s <= t) ? __expf(sAcum[t] - sAcum[s]) : 0.f;
        sM[t * LD + s] = (bf16_t)(acc[j][v] * lm);
      }
  }
  __syncthreads();

  // GEMM 2: Y_diag[t,p] = sum_s M[t,s] X[s,p] -> y (f32)
  {
    v8f acc[4] = {};
#pragma unroll
    for (int kk = 0; kk < 64; kk += 32) {
      v16bf aF = ldsA_frag(&sM[m0 * LD + kk], LD);
#pragma unroll
      for (int j = 0; j < 4; ++j)
        acc[j] = wmma_bf16(aF, ldsB_frag(&sXT[(j * 16) * LD + kk], LD), acc[j]);
    }
#pragma unroll
    for (int j = 0; j < 4; ++j)
#pragma unroll
      for (int v = 0; v < 8; ++v) {
        const int t = m0 + v + 8 * half;
        const int p = j * 16 + nn;
        y[(tok0 + t) * DINNER + h * HEADDIM + p] = acc[j][v];
      }
  }

  // GEMM 3: state[p,n] = sum_t decay[t] X[t,p] B[t,n] -> states (f32)
  {
    v8f acc[4] = {};
#pragma unroll
    for (int kk = 0; kk < 64; kk += 32) {
      v16bf aF = ldsA_frag(&sXD[m0 * LD + kk], LD);
#pragma unroll
      for (int j = 0; j < 4; ++j)
        acc[j] = wmma_bf16(aF, ldsB_frag(&sBT[(j * 16) * LD + kk], LD), acc[j]);
    }
    float* st = states + (((size_t)(b * 64 + h) * 64 + c) * 64) * 64;
#pragma unroll
    for (int j = 0; j < 4; ++j)
#pragma unroll
      for (int v = 0; v < 8; ++v) {
        const int p = m0 + v + 8 * half;
        const int n = j * 16 + nn;
        st[p * 64 + n] = acc[j][v];
      }
  }
}

// -------------- inter-chunk recurrence (in-place over states) --------------
// states[b,h,c] holds local chunk state on entry; state ENTERING chunk c on exit.
__global__ __launch_bounds__(256) void scan_seq_kernel(
    const float* __restrict__ a_dt, float* states) {
  const int bh = blockIdx.x;
  const int h = bh & 63, b = bh >> 6;
  const int tid = threadIdx.x;
  __shared__ float sSum[64];
  if (tid < 64) {                       // chunk a-sum
    float s = 0.f;
    const float* ap = a_dt + ((size_t)b * L_SEQ + tid * CHUNKT) * NHEADS + h;
    for (int t = 0; t < 64; ++t) s += ap[(size_t)t * NHEADS];
    sSum[tid] = s;
  }
  __syncthreads();
  float prev[16];
#pragma unroll
  for (int i = 0; i < 16; ++i) prev[i] = 0.f;
  float* base = states + ((size_t)(b * 64 + h) * 64) * 4096 + tid * 16;
  for (int cI = 0; cI < 64; ++cI) {
    float* p = base + (size_t)cI * 4096;
    float loc[16];
#pragma unroll
    for (int i = 0; i < 16; ++i) loc[i] = p[i];
#pragma unroll
    for (int i = 0; i < 16; ++i) p[i] = prev[i];
    const float d = __expf(sSum[cI]);
#pragma unroll
    for (int i = 0; i < 16; ++i) prev[i] = d * prev[i] + loc[i];
  }
}

// ----------- inter-chunk output: Y += exp(Acum)·(C·state^T) + D·x ----------
__global__ __launch_bounds__(128) void scan_inter_kernel(
    const float* __restrict__ xbc, const float* __restrict__ a_dt,
    const float* __restrict__ states, const float* __restrict__ Dp,
    float* __restrict__ y) {
  constexpr int LD = 72;
  __shared__ bf16_t sC[64 * LD];      // C [t][n]
  __shared__ bf16_t sS[64 * LD];      // state [p][n] (B: col=p, k=n)
  __shared__ float  sAcum[64];

  const int bx = blockIdx.x;
  const int h = bx & 63, c = (bx >> 6) & 63, b = bx >> 12;
  const int tid = threadIdx.x, lane = tid & 31, wave = tid >> 5;
  const size_t tok0 = (size_t)b * L_SEQ + c * CHUNKT;

  if (tid < 64) sAcum[tid] = a_dt[(tok0 + tid) * NHEADS + h];
  __syncthreads();
  if (tid == 0) {
    float s = 0.f;
    for (int t = 0; t < 64; ++t) { s += sAcum[t]; sAcum[t] = s; }
  }
  __syncthreads();

  const float* st = states + (((size_t)(b * 64 + h) * 64 + c) * 64) * 64;
  for (int i = tid; i < 64 * 64; i += 128) {
    const int t = i >> 6, n = i & 63;
    sC[t * LD + n] = (bf16_t)xbc[(tok0 + t) * DXBC + DINNER + DSTATE + n];
    sS[t * LD + n] = (bf16_t)st[t * 64 + n];    // t plays the role of p here
  }
  __syncthreads();

  const int m0 = wave * 16;
  const int half = lane >> 4, nn = lane & 15;
  const float Dh = Dp[h];

  v8f acc[4] = {};
#pragma unroll
  for (int kk = 0; kk < 64; kk += 32) {
    v16bf aF = ldsA_frag(&sC[m0 * LD + kk], LD);
#pragma unroll
    for (int j = 0; j < 4; ++j)
      acc[j] = wmma_bf16(aF, ldsB_frag(&sS[(j * 16) * LD + kk], LD), acc[j]);
  }
#pragma unroll
  for (int j = 0; j < 4; ++j)
#pragma unroll
    for (int v = 0; v < 8; ++v) {
      const int t = m0 + v + 8 * half;
      const int p = j * 16 + nn;
      const size_t yi = (tok0 + t) * DINNER + h * HEADDIM + p;
      const float xraw = xbc[(tok0 + t) * DXBC + h * HEADDIM + p];
      y[yi] = y[yi] + acc[j][v] * __expf(sAcum[t]) + Dh * xraw;
    }
}

// ------------- gate (y * silu(z)) + grouped RMSNorm -> bf16 ----------------
__global__ __launch_bounds__(256) void gate_norm_kernel(
    const float* __restrict__ y, const float* __restrict__ zxbcdt,
    const float* __restrict__ norm_w, bf16_t* __restrict__ yg) {
  const int bt = blockIdx.x;
  const int tid = threadIdx.x;
  const float* yr = y + (size_t)bt * DINNER;
  const float* zr = zxbcdt + (size_t)bt * DPROJ;       // z = first DINNER cols
  float g[16];
  float ss = 0.f;
#pragma unroll
  for (int i = 0; i < 16; ++i) {
    const int ch = tid + i * 256;
    const float z = zr[ch];
    const float gg = yr[ch] * (z / (1.f + __expf(-z)));
    g[i] = gg;
    ss += gg * gg;
  }
  __shared__ float red[256];
  red[tid] = ss;
  __syncthreads();
  for (int s = 128; s > 0; s >>= 1) {
    if (tid < s) red[tid] += red[tid + s];
    __syncthreads();
  }
  const float scale = rsqrtf(red[0] / (float)DINNER + 1e-5f);
#pragma unroll
  for (int i = 0; i < 16; ++i) {
    const int ch = tid + i * 256;
    yg[(size_t)bt * DINNER + ch] = (bf16_t)(g[i] * scale * norm_w[ch]);
  }
}

// ---------------------------------------------------------------------------
extern "C" void kernel_launch(void* const* d_in, const int* in_sizes, int n_in,
                              void* d_out, int out_size, void* d_ws,
                              size_t ws_size, hipStream_t stream) {
  const float* u       = (const float*)d_in[0];
  const float* W_in    = (const float*)d_in[1];
  const float* conv_w  = (const float*)d_in[2];
  const float* conv_b  = (const float*)d_in[3];
  const float* dt_bias = (const float*)d_in[4];
  const float* A_log   = (const float*)d_in[5];
  const float* D_param = (const float*)d_in[6];
  const float* norm_w  = (const float*)d_in[7];
  const float* W_out   = (const float*)d_in[8];
  float* out = (float*)d_out;

  // workspace layout (bytes, 256-aligned)
  char* ws = (char*)d_ws;
  size_t off = 0;
  auto alloc = [&](size_t bytes) {
    void* p = ws + off;
    off += (bytes + 255) & ~(size_t)255;
    return p;
  };
  bf16_t* u_bf    = (bf16_t*)alloc((size_t)NTOK * DMODEL * 2);
  bf16_t* win_bf  = (bf16_t*)alloc((size_t)DPROJ * DMODEL * 2);
  bf16_t* wout_bf = (bf16_t*)alloc((size_t)DMODEL * DINNER * 2);
  float*  zxbcdt  = (float*)alloc((size_t)NTOK * DPROJ * 4);
  float*  xbcc    = (float*)alloc((size_t)NTOK * DXBC * 4);
  float*  dt_soft = (float*)alloc((size_t)NTOK * NHEADS * 4);
  float*  a_dt    = (float*)alloc((size_t)NTOK * NHEADS * 4);
  float*  states  = (float*)alloc((size_t)NB_B * NHEADS * NCHUNK * 4096 * 4);
  float*  ybuf    = (float*)alloc((size_t)NTOK * DINNER * 4);
  bf16_t* ygate   = (bf16_t*)alloc((size_t)NTOK * DINNER * 2);

  // 1) f32 -> bf16 operand conversion
  {
    unsigned n;
    n = (unsigned)NTOK * DMODEL;
    f32_to_bf16_kernel<<<(n + 255) / 256, 256, 0, stream>>>(u, u_bf, n);
    n = (unsigned)DPROJ * DMODEL;
    f32_to_bf16_kernel<<<(n + 255) / 256, 256, 0, stream>>>(W_in, win_bf, n);
    n = (unsigned)DMODEL * DINNER;
    f32_to_bf16_kernel<<<(n + 255) / 256, 256, 0, stream>>>(W_out, wout_bf, n);
  }

  // 2) in_proj: zxbcdt[8192,8448] = u[8192,2048] @ W_in^T
  gemm_bf16_wmma_kernel<<<dim3(DPROJ / 128, NTOK / 128), 256, 0, stream>>>(
      u_bf, win_bf, zxbcdt, NTOK, DPROJ, DMODEL);

  // 3) conv + SiLU, 4) dt softplus / A*dt
  {
    unsigned n = (unsigned)NTOK * DXBC;
    conv_silu_kernel<<<(n + 255) / 256, 256, 0, stream>>>(zxbcdt, conv_w,
                                                          conv_b, xbcc);
    n = (unsigned)NTOK * NHEADS;
    dt_kernel<<<(n + 255) / 256, 256, 0, stream>>>(zxbcdt, dt_bias, A_log,
                                                   dt_soft, a_dt);
  }

  // 5) intra-chunk scan (Y_diag + local states), 6) inter-chunk recurrence,
  // 7) inter-chunk output (+ D*x)
  scan_intra_kernel<<<NB_B * NCHUNK * NHEADS, 128, 0, stream>>>(
      xbcc, dt_soft, a_dt, ybuf, states);
  scan_seq_kernel<<<NB_B * NHEADS, 256, 0, stream>>>(a_dt, states);
  scan_inter_kernel<<<NB_B * NCHUNK * NHEADS, 128, 0, stream>>>(
      xbcc, a_dt, states, D_param, ybuf);

  // 8) gate + RMSNorm -> bf16
  gate_norm_kernel<<<NTOK, 256, 0, stream>>>(ybuf, zxbcdt, norm_w, ygate);

  // 9) out_proj: out[8192,2048] = ygate[8192,4096] @ W_out^T
  gemm_bf16_wmma_kernel<<<dim3(DMODEL / 128, NTOK / 128), 256, 0, stream>>>(
      ygate, wout_bf, out, NTOK, DMODEL, DINNER);
}
